// SwinBlock3D_21981642621143
// MI455X (gfx1250) — matrix operand order repaired and checked by reading, not compiled, not run on hardware
//
#include <hip/hip_runtime.h>
#include <hip/hip_bf16.h>

typedef __attribute__((ext_vector_type(16))) _Float16 v16h;
typedef __attribute__((ext_vector_type(8)))  float    v8f;

#define NH 4
#define HD 32
#define SCALE 0.17677669529663689f

// ---------------------------------------------------------------------------
// WMMA helpers (gfx1250, wave32): D(16x16,f32) = A(16x32,f16) x B(32x16,f16) + C
// ---------------------------------------------------------------------------
__device__ __forceinline__ v8f wmma_f16(v16h a, v16h b, v8f c) {
    return __builtin_amdgcn_wmma_f32_16x16x32_f16(
        /*neg_a=*/false, a, /*neg_b=*/false, b,
        /*c_mod=*/(short)0, c, /*reuse_a=*/false, /*reuse_b=*/false);
}

// Async DMA: 16B global -> 16B LDS per lane, tracked by ASYNCcnt (no VGPR data).
__device__ __forceinline__ void async_load_b128(unsigned lds_byte_off, const void* gaddr) {
    asm volatile("global_load_async_to_lds_b128 %0, %1, off"
                 :: "v"(lds_byte_off), "v"(gaddr)
                 : "memory");
}
__device__ __forceinline__ void wait_async_zero() {
    asm volatile("s_wait_asynccnt 0" ::: "memory");
}

// A fragment from row-major LDS matrix A[M][K] (stride ld halfs).
__device__ __forceinline__ v16h a_frag(const _Float16* A, int ld, int row0, int k0) {
    const int l = threadIdx.x & 31;
    const _Float16* p = A + (size_t)(row0 + (l & 15)) * ld + k0 + ((l & 16) ? 16 : 0);
    v16h f;
#pragma unroll
    for (int e = 0; e < 8; ++e)  f[e] = p[e];        // k = khi + e
#pragma unroll
    for (int e = 8; e < 16; ++e) f[e] = p[e];        // k = khi + e
    return f;
}

// B fragment where B(k,n) is stored "n-major": Bt[n*ld + k]  (i.e. B^T row-major).
__device__ __forceinline__ v16h b_frag_nk(const _Float16* Bt, int ld, int k0, int n0) {
    const int l = threadIdx.x & 31;
    const _Float16* p = Bt + (size_t)(n0 + (l & 15)) * ld + k0 + ((l & 16) ? 16 : 0);
    v16h f;
#pragma unroll
    for (int j = 0; j < 16; ++j) f[j] = p[j];
    return f;
}

// B fragment from pre-packed weights: tile (kt,nt) = 32 lanes x 16 halfs contiguous.
__device__ __forceinline__ v16h b_frag_packed(const _Float16* P, int ktiles, int kt, int nt) {
    const int l = threadIdx.x & 31;
    const _Float16* p = P + (((size_t)(nt * ktiles + kt)) << 9) + (l << 4);
    v16h f;
#pragma unroll
    for (int j = 0; j < 16; ++j) f[j] = p[j];
    return f;
}

// ---------------------------------------------------------------------------
// Pre-pass: convert fp32 weight W[K][N] into f16 WMMA-B-fragment-packed layout.
// ---------------------------------------------------------------------------
__global__ void pack_w(const float* __restrict__ W, _Float16* __restrict__ P, int K, int N) {
    int idx = blockIdx.x * 256 + threadIdx.x;
    if (idx >= K * N) return;
    int tile = idx >> 9, lane = (idx >> 4) & 31, j = idx & 15;
    int ktiles = K >> 5;
    int nt = tile / ktiles, kt = tile - nt * ktiles;
    int k = kt * 32 + ((lane & 16) ? 16 : 0) + j;
    int n = nt * 16 + (lane & 15);
    P[idx] = (_Float16)W[(size_t)k * N + n];
}

// ---------------------------------------------------------------------------
// Fused Swin3D block: 1 workgroup == 1 window (64 tokens x 128 ch), 8 waves.
// ---------------------------------------------------------------------------
// XW is channel-major [c][n], stride 72 floats (288B rows: 16B-aligned for async
// b128 DMA; the 4-token float4 along W lands as 16 contiguous LDS bytes).
#define XW_LD    72
#define OFF_XW   0                       // float [128][72]  = 36864 B (residual)
#define OFF_HH   36864                   // f16   [64][130]  = 16640 B (activations)
#define OFF_U    53504                   // union region
#define U_QH     0                       // f16 [64][130]  q (pre-scaled)
#define U_KH     16640                   // f16 [64][130]  k
#define U_VT     33280                   // f16 [128][66]  v transposed
#define U_ATT    50176                   // f16 [4][64][66] attn probs
#define U_RPB    83968                   // float [343][4] staged rel-pos table
#define U_END    89456
#define SMEM_BYTES (OFF_U + U_END)       // 142960 B -> 2 workgroups per 320KB WGP

__global__ __launch_bounds__(256) void swin3d_fused(
    const float* __restrict__ x,
    const _Float16* __restrict__ Pqkv, const _Float16* __restrict__ Pproj,
    const _Float16* __restrict__ Pw1,  const _Float16* __restrict__ Pw2,
    const float* __restrict__ qkv_b, const float* __restrict__ proj_b,
    const float* __restrict__ rpb,
    const float* __restrict__ ln1g, const float* __restrict__ ln1b,
    const float* __restrict__ ln2g, const float* __restrict__ ln2b,
    const float* __restrict__ mb1, const float* __restrict__ mb2,
    float* __restrict__ out)
{
    extern __shared__ char smem[];
    float*    XW  = (float*)(smem + OFF_XW);            // [c][n] stride XW_LD
    _Float16* HH  = (_Float16*)(smem + OFF_HH);         // stride 130
    _Float16* QH  = (_Float16*)(smem + OFF_U + U_QH);   // stride 130
    _Float16* KH  = (_Float16*)(smem + OFF_U + U_KH);   // stride 130
    _Float16* VT  = (_Float16*)(smem + OFF_U + U_VT);   // stride 66, [c][m]
    _Float16* ATT = (_Float16*)(smem + OFF_U + U_ATT);  // stride 66, [h][n][m]
    float*    RPB = (float*)(smem + OFF_U + U_RPB);
    _Float16* M1  = (_Float16*)(smem + OFF_U + U_QH);   // stride 514 (aliases attn bufs)

    const int tid  = threadIdx.x;
    const int lane = tid & 31;
    // Force wave id scalar-uniform so tile loops & epilogue branches lower to SALU.
    const int wave = __builtin_amdgcn_readfirstlane(tid >> 5);

    // window decode: 2048 windows/batch = 8(z) * 16(y) * 16(x)
    const int wid = blockIdx.x;
    const int b  = wid >> 11;
    const int r  = wid & 2047;
    const int z0 = (r >> 8) * 4;
    const int y0 = ((r >> 4) & 15) * 4;
    const int x0 = (r & 15) * 4;

    // ---- async DMA window -> XW[c][n] (b128 per lane, no VGPR round-trip) ----
    for (int it = tid; it < 2048; it += 256) {
        int c  = it >> 4;
        int wz = (it >> 2) & 3;
        int wy = it & 3;
        const float* g = x +
            ((((size_t)b * 128 + c) * 32 + (z0 + wz)) * 64 + (y0 + wy)) * 64 + x0;
        unsigned lds = (unsigned)OFF_XW + (unsigned)((c * XW_LD + wz * 16 + wy * 4) * 4);
        async_load_b128(lds, g);
    }
    // ---- async DMA rel-pos table (343*4 floats = 343 x b128) ----
    for (int it = tid; it < 343; it += 256) {
        async_load_b128((unsigned)(OFF_U + U_RPB) + (unsigned)(it * 16),
                        (const char*)rpb + it * 16);
    }
    wait_async_zero();
    __syncthreads();

    // ---- LN1 -> HH (f16). 4 lanes per token row, shfl reduce. ----
    {
        int n = tid >> 2, qq = tid & 3;
        float s = 0.f, s2 = 0.f;
#pragma unroll
        for (int i = 0; i < 32; ++i) {
            float v = XW[(qq * 32 + i) * XW_LD + n]; s += v; s2 += v * v;
        }
        s  += __shfl_xor(s, 1);  s  += __shfl_xor(s, 2);
        s2 += __shfl_xor(s2, 1); s2 += __shfl_xor(s2, 2);
        float mu = s * (1.f / 128.f);
        float rs = rsqrtf(s2 * (1.f / 128.f) - mu * mu + 1e-5f);
#pragma unroll
        for (int i = 0; i < 32; ++i) {
            int c = qq * 32 + i;
            HH[n * 130 + c] = (_Float16)((XW[c * XW_LD + n] - mu) * rs * ln1g[c] + ln1b[c]);
        }
    }
    __syncthreads();

    // ---- QKV: (64x128) @ (128x384). 96 tiles, 12 per wave. ----
    for (int t = wave; t < 96; t += 8) {
        int mt = t & 3, nt = t >> 2;                 // uniform per wave
        __builtin_prefetch(Pqkv + (((size_t)(nt * 4)) << 9), 0, 3);
        v8f acc = {};
#pragma unroll
        for (int kt = 0; kt < 4; ++kt)
            acc = wmma_f16(a_frag(HH, 130, mt * 16, kt * 32),
                           b_frag_packed(Pqkv, 4, kt, nt), acc);
        int col = nt * 16 + (lane & 15);
        int mb  = mt * 16 + ((lane & 16) ? 8 : 0);
        float bias = qkv_b[col];
        if (nt < 8) {                                // q tile (uniform branch)
#pragma unroll
            for (int rr = 0; rr < 8; ++rr)
                QH[(mb + rr) * 130 + col] = (_Float16)((acc[rr] + bias) * SCALE);
        } else if (nt < 16) {                        // k tile
            int ck = col - 128;
#pragma unroll
            for (int rr = 0; rr < 8; ++rr)
                KH[(mb + rr) * 130 + ck] = (_Float16)(acc[rr] + bias);
        } else {                                     // v tile -> transposed
            int cv = col - 256;
#pragma unroll
            for (int rr = 0; rr < 8; ++rr)
                VT[cv * 66 + (mb + rr)] = (_Float16)(acc[rr] + bias);
        }
    }
    __syncthreads();

    // ---- Attention scores + softmax in registers. 16 units (head x rowblock). ----
    for (int u = wave; u < 16; u += 8) {
        int hd = u >> 2, mt = u & 3;
        v16h aq = a_frag(QH, 130, mt * 16, hd * HD);   // reused across 4 column tiles
        v8f s[4];
#pragma unroll
        for (int ct = 0; ct < 4; ++ct) {
            v8f z = {};
            s[ct] = wmma_f16(aq, b_frag_nk(KH + hd * HD, 130, 0, ct * 16), z);
        }
        int mb   = mt * 16 + ((lane & 16) ? 8 : 0);
        int mcol = lane & 15;
#pragma unroll
        for (int rr = 0; rr < 8; ++rr) {
            int n = mb + rr;
            int zn = n >> 4, yn = (n >> 2) & 3, xn = n & 3;
            float vals[4];
            float mx = -3.0e38f;
#pragma unroll
            for (int ct = 0; ct < 4; ++ct) {
                int m = ct * 16 + mcol;
                int zm = m >> 4, ym = (m >> 2) & 3, xm = m & 3;
                int ridx = (zn - zm + 3) * 49 + (yn - ym + 3) * 7 + (xn - xm + 3);
                float v = s[ct][rr] + RPB[ridx * NH + hd];
                vals[ct] = v;
                mx = fmaxf(mx, v);
            }
            mx = fmaxf(mx, __shfl_xor(mx, 1));
            mx = fmaxf(mx, __shfl_xor(mx, 2));
            mx = fmaxf(mx, __shfl_xor(mx, 4));
            mx = fmaxf(mx, __shfl_xor(mx, 8));
            float sum = 0.f;
#pragma unroll
            for (int ct = 0; ct < 4; ++ct) { vals[ct] = __expf(vals[ct] - mx); sum += vals[ct]; }
            sum += __shfl_xor(sum, 1);
            sum += __shfl_xor(sum, 2);
            sum += __shfl_xor(sum, 4);
            sum += __shfl_xor(sum, 8);
            float inv = 1.f / sum;
#pragma unroll
            for (int ct = 0; ct < 4; ++ct)
                ATT[((size_t)hd * 64 + n) * 66 + ct * 16 + mcol] = (_Float16)(vals[ct] * inv);
        }
    }
    __syncthreads();

    // ---- attn @ V -> HH (f16). 32 tiles, 4 per wave. ----
    for (int t = wave; t < 32; t += 8) {
        int mt = t & 3, ct = t >> 2;
        int hd = ct >> 1;
        v8f acc = {};
#pragma unroll
        for (int kt = 0; kt < 2; ++kt)
            acc = wmma_f16(a_frag(ATT + (size_t)hd * 64 * 66, 66, mt * 16, kt * 32),
                           b_frag_nk(VT, 66, kt * 32, ct * 16), acc);
        int col = ct * 16 + (lane & 15);
        int mb  = mt * 16 + ((lane & 16) ? 8 : 0);
#pragma unroll
        for (int rr = 0; rr < 8; ++rr)
            HH[(mb + rr) * 130 + col] = (_Float16)acc[rr];
    }
    __syncthreads();

    // ---- proj: (64x128)@(128x128), + bias, residual into XW[c][n]. ----
    for (int t = wave; t < 32; t += 8) {
        int mt = t & 3, nt = t >> 2;
        __builtin_prefetch(Pproj + (((size_t)(nt * 4)) << 9), 0, 3);
        v8f acc = {};
#pragma unroll
        for (int kt = 0; kt < 4; ++kt)
            acc = wmma_f16(a_frag(HH, 130, mt * 16, kt * 32),
                           b_frag_packed(Pproj, 4, kt, nt), acc);
        int col = nt * 16 + (lane & 15);
        int mb  = mt * 16 + ((lane & 16) ? 8 : 0);
        float bias = proj_b[col];
        float* xr = XW + col * XW_LD + mb;           // 8 contiguous floats
#pragma unroll
        for (int rr = 0; rr < 8; ++rr)
            xr[rr] += acc[rr] + bias;
    }
    __syncthreads();

    // ---- LN2 -> HH ----
    {
        int n = tid >> 2, qq = tid & 3;
        float s = 0.f, s2 = 0.f;
#pragma unroll
        for (int i = 0; i < 32; ++i) {
            float v = XW[(qq * 32 + i) * XW_LD + n]; s += v; s2 += v * v;
        }
        s  += __shfl_xor(s, 1);  s  += __shfl_xor(s, 2);
        s2 += __shfl_xor(s2, 1); s2 += __shfl_xor(s2, 2);
        float mu = s * (1.f / 128.f);
        float rs = rsqrtf(s2 * (1.f / 128.f) - mu * mu + 1e-5f);
#pragma unroll
        for (int i = 0; i < 32; ++i) {
            int c = qq * 32 + i;
            HH[n * 130 + c] = (_Float16)((XW[c * XW_LD + n] - mu) * rs * ln2g[c] + ln2b[c]);
        }
    }
    __syncthreads();

    // ---- MLP1: (64x128)@(128x512) + bias + exact GELU -> M1 (f16). ----
    for (int t = wave; t < 128; t += 8) {
        int mt = t & 3, nt = t >> 2;
        __builtin_prefetch(Pw1 + (((size_t)(nt * 4)) << 9), 0, 3);
        v8f acc = {};
#pragma unroll
        for (int kt = 0; kt < 4; ++kt)
            acc = wmma_f16(a_frag(HH, 130, mt * 16, kt * 32),
                           b_frag_packed(Pw1, 4, kt, nt), acc);
        int col = nt * 16 + (lane & 15);
        int mb  = mt * 16 + ((lane & 16) ? 8 : 0);
        float bias = mb1[col];
#pragma unroll
        for (int rr = 0; rr < 8; ++rr) {
            float v = acc[rr] + bias;
            float g = 0.5f * v * (1.0f + erff(v * 0.70710678118f));
            M1[(size_t)(mb + rr) * 514 + col] = (_Float16)g;
        }
    }
    __syncthreads();

    // ---- MLP2: (64x512)@(512x128) + bias + residual -> global out. ----
    for (int t = wave; t < 32; t += 8) {
        int mt = t & 3, nt = t >> 2;
        __builtin_prefetch(Pw2 + (((size_t)(nt * 16)) << 9), 0, 3);
        v8f acc = {};
#pragma unroll
        for (int kt = 0; kt < 16; ++kt)
            acc = wmma_f16(a_frag(M1, 514, mt * 16, kt * 32),
                           b_frag_packed(Pw2, 16, kt, nt), acc);
        int col = nt * 16 + (lane & 15);
        int mb  = mt * 16 + ((lane & 16) ? 8 : 0);
        float bias = mb2[col];
        const float* xr = XW + col * XW_LD + mb;     // 8 contiguous floats
#pragma unroll
        for (int rr = 0; rr < 8; ++rr) {
            int n = mb + rr;
            float v = acc[rr] + bias + xr[rr];
            int wz = n >> 4, wy = (n >> 2) & 3, wx = n & 3;
            out[((((size_t)b * 128 + col) * 32 + (z0 + wz)) * 64 + (y0 + wy)) * 64 + (x0 + wx)] = v;
        }
    }
}

extern "C" void kernel_launch(void* const* d_in, const int* in_sizes, int n_in,
                              void* d_out, int out_size, void* d_ws, size_t ws_size,
                              hipStream_t stream) {
    const float* x      = (const float*)d_in[0];
    const float* qkv_w  = (const float*)d_in[1];
    const float* qkv_b  = (const float*)d_in[2];
    const float* proj_w = (const float*)d_in[3];
    const float* proj_b = (const float*)d_in[4];
    const float* rpb    = (const float*)d_in[5];
    const float* ln1g   = (const float*)d_in[6];
    const float* ln1b   = (const float*)d_in[7];
    const float* ln2g   = (const float*)d_in[8];
    const float* ln2b   = (const float*)d_in[9];
    const float* mlp_w1 = (const float*)d_in[10];
    const float* mlp_b1 = (const float*)d_in[11];
    const float* mlp_w2 = (const float*)d_in[12];
    const float* mlp_b2 = (const float*)d_in[13];

    // f16 fragment-packed weights in workspace (384 KB total)
    _Float16* P     = (_Float16*)d_ws;
    _Float16* Pqkv  = P;                 // 128x384 = 49152
    _Float16* Pproj = P + 49152;         // 128x128 = 16384
    _Float16* Pw1   = P + 65536;         // 128x512 = 65536
    _Float16* Pw2   = P + 131072;        // 512x128 = 65536

    pack_w<<<192, 256, 0, stream>>>(qkv_w,  Pqkv,  128, 384);
    pack_w<<<64,  256, 0, stream>>>(proj_w, Pproj, 128, 128);
    pack_w<<<256, 256, 0, stream>>>(mlp_w1, Pw1,   128, 512);
    pack_w<<<256, 256, 0, stream>>>(mlp_w2, Pw2,   512, 128);

    // 4096 windows, one workgroup (8 wave32) each, ~140 KB LDS (2 WGs / WGP).
    swin3d_fused<<<4096, 256, SMEM_BYTES, stream>>>(
        x, Pqkv, Pproj, Pw1, Pw2,
        qkv_b, proj_b, rpb, ln1g, ln1b, ln2g, ln2b, mlp_b1, mlp_b2,
        (float*)d_out);
}
